// VarMaskedRNNBase_73924977098878
// MI455X (gfx1250) — matrix operand order repaired and checked by reading, not compiled, not run on hardware
//
#include <hip/hip_runtime.h>

// ---------------------------------------------------------------------------
// Bidirectional 2-layer LSTM with variational dropout (VarMaskedRNN) for
// MI455X / gfx1250.  Strategy:
//   * bf16 WMMA (v_wmma_f32_16x16x32_bf16) for both the big input-projection
//     GEMMs (gx = x @ w_ih^T + b, precomputed for all T) and the per-step
//     hidden GEMM ((h*noise) @ w_hh^T).
//   * gx GEMM: each wave computes a 32(M)x128(N) strip -> every B fragment is
//     reused by two WMMAs (20 b128 loads per 16 WMMAs).
//   * Per-step kernel: each wave owns a 16-wide hidden slice and computes all
//     4 gate tiles for all 32 batch rows -> gates combine entirely in
//     registers, then cell update + masking + bf16 re-quantization of h*noise.
//   * h*noise bf16 buffer double-buffered by time parity (read by every
//     workgroup of a step while being rewritten).
//   * All weights in bf16 (~21 MB) stay resident in the 192 MB L2 across the
//     1024 serial steps.
// ---------------------------------------------------------------------------

typedef __bf16 bf16_t;
typedef __attribute__((ext_vector_type(16))) __bf16 v16bf;
typedef __attribute__((ext_vector_type(8)))  __bf16 v8bf;
typedef __attribute__((ext_vector_type(8)))  float  v8f;

constexpr int BB = 32;    // batch
constexpr int TT = 512;   // time steps
constexpr int DD = 512;   // input feature dim (layer 0)
constexpr int HH = 512;   // hidden size
constexpr int GG = 2048;  // 4*H gates

__device__ __forceinline__ v8f vzero8() {
  v8f z;
#pragma unroll
  for (int j = 0; j < 8; ++j) z[j] = 0.0f;
  return z;
}

__device__ __forceinline__ v8f wmma_bf16(v16bf a, v16bf b, v8f c) {
  // D = A(16x32 bf16) * B(32x16 bf16) + C(16x16 f32)
  return __builtin_amdgcn_wmma_f32_16x16x32_bf16(
      /*neg_a=*/false, a, /*neg_b=*/false, b,
      /*c_mod=*/(short)0, c, /*reuse_a=*/false, /*reuse_b=*/false);
}

// A fragment: 16x32 tile, source row-major bf16 with leading dim lda.
// ISA layout (16-bit A 16x32): lanes 0-15 hold K = [k0..k0+7] and [k0+16..k0+23],
// lanes 16-31 hold K = [k0+8..k0+15] and [k0+24..k0+31]; M = lane%16 (row passed in).
__device__ __forceinline__ v16bf load_a_frag(const bf16_t* __restrict__ A,
                                             int lda, int row, int k0, int lane) {
  const int sel = lane >> 4;
  const bf16_t* p = A + (size_t)row * lda + k0 + 8 * sel;
  v8bf lo = *(const v8bf*)(p);
  v8bf hi = *(const v8bf*)(p + 16);
  v16bf a;
#pragma unroll
  for (int j = 0; j < 8; ++j) { a[j] = lo[j]; a[j + 8] = hi[j]; }
  return a;
}

// B fragment: B[k][n] = W[n][k]  (W row-major, leading dim ldk).
// ISA layout: lane holds N = n0 + lane%16, K = k0 + 8*(lane/16) + [0..7] and k0 + 16 + 8*(lane/16) + [0..7].
__device__ __forceinline__ v16bf load_b_frag(const bf16_t* __restrict__ W,
                                             int ldk, int n0, int k0, int lane) {
  const int n = n0 + (lane & 15);
  const bf16_t* p = W + (size_t)n * ldk + k0 + 8 * (lane >> 4);
  v8bf lo = *(const v8bf*)(p);
  v8bf hi = *(const v8bf*)(p + 16);
  v16bf b;
#pragma unroll
  for (int j = 0; j < 8; ++j) { b[j] = lo[j]; b[j + 8] = hi[j]; }
  return b;
}

// ---------------------------------------------------------------------------
// Conversion kernels
// ---------------------------------------------------------------------------
__global__ void cvt_kernel(const float* __restrict__ src, bf16_t* __restrict__ dst, int n) {
  int i = blockIdx.x * blockDim.x + threadIdx.x;
  if (i < n) dst[i] = (bf16_t)src[i];
}

// x: [B,T,D] fp32  ->  xb: [T,B,D] bf16 (time-major rows for the GEMM A matrix)
__global__ void cvt_x_kernel(const float* __restrict__ x, bf16_t* __restrict__ xb) {
  size_t i = (size_t)blockIdx.x * blockDim.x + threadIdx.x;
  if (i >= (size_t)BB * TT * DD) return;
  int d = (int)(i % DD);
  size_t bt = i / DD;
  int t = (int)(bt % TT);
  int b = (int)(bt / TT);
  xb[((size_t)t * BB + b) * DD + d] = (bf16_t)x[i];
}

// ---------------------------------------------------------------------------
// gx GEMM:  gx[dir][t_scan*B+b][g] = sum_k A[src_row][k] * W[g][k] + bi[g]+bh[g]
// dir==1 reads A rows time-reversed (scan order).  M = T*B, N = 2048, K in arg.
// Block: 256 threads (8 waves); wave -> one 32(M) x 128(N) output strip
// (2 M-tiles x 8 N-tiles = 16 accumulators; B fragments reused across M-tiles).
// Grid: (N/128, M/(32*8), 2 dirs)
// ---------------------------------------------------------------------------
__global__ __launch_bounds__(256)
void gemm_gx_kernel(const bf16_t* __restrict__ A,
                    const bf16_t* __restrict__ w0, const bf16_t* __restrict__ w1,
                    const float* __restrict__ bi0, const float* __restrict__ bh0,
                    const float* __restrict__ bi1, const float* __restrict__ bh1,
                    float* __restrict__ gx, int K) {
  const int tid  = threadIdx.x;
  const int lane = tid & 31;
  const int wave = tid >> 5;
  const int dir  = blockIdx.z;
  const int m0   = (blockIdx.y * 8 + wave) * 32;
  const int n0   = blockIdx.x * 128;

  const bf16_t* W  = dir ? w1 : w0;
  const float*  bi = dir ? bi1 : bi0;
  const float*  bh = dir ? bh1 : bh0;
  float* gxd = gx + (size_t)dir * TT * BB * GG;

  // per-lane A source rows for the two M-tiles (scan order -> actual time,
  // reversed for dir 1).  B == 32, m0 is a multiple of 32, so each M-tile
  // pair covers exactly one timestep.
  const int ml0 = m0 + (lane & 15);        // M-tile 0 row (scan order)
  const int ml1 = ml0 + 16;                // M-tile 1 row
  const int t0 = ml0 / BB, b0 = ml0 % BB;
  const int t1 = ml1 / BB, b1 = ml1 % BB;
  const int srow0 = dir ? ((TT - 1 - t0) * BB + b0) : ml0;
  const int srow1 = dir ? ((TT - 1 - t1) * BB + b1) : ml1;

  v8f acc[8][2];
#pragma unroll
  for (int i = 0; i < 8; ++i) { acc[i][0] = vzero8(); acc[i][1] = vzero8(); }

  for (int k0 = 0; k0 < K; k0 += 32) {
    v16bf a0 = load_a_frag(A, K, srow0, k0, lane);
    v16bf a1 = load_a_frag(A, K, srow1, k0, lane);
#pragma unroll
    for (int i = 0; i < 8; ++i) {
      v16bf bf = load_b_frag(W, K, n0 + 16 * i, k0, lane);
      acc[i][0] = wmma_bf16(a0, bf, acc[i][0]);
      acc[i][1] = wmma_bf16(a1, bf, acc[i][1]);
    }
  }

  const int nl = lane & 15, hb = lane >> 4;
#pragma unroll
  for (int i = 0; i < 8; ++i) {
    const int n = n0 + 16 * i + nl;
    const float biasv = bi[n] + bh[n];
#pragma unroll
    for (int mt = 0; mt < 2; ++mt) {
#pragma unroll
      for (int r = 0; r < 8; ++r) {
        const int mr = m0 + mt * 16 + r + 8 * hb;
        gxd[(size_t)mr * GG + n] = acc[i][mt][r] + biasv;
      }
    }
  }
}

// ---------------------------------------------------------------------------
// State init: zero h, c (fp32) and both parities of the bf16 h*noise buffer.
// ---------------------------------------------------------------------------
__global__ void lstm_init_kernel(float* __restrict__ hst, float* __restrict__ cst,
                                 bf16_t* __restrict__ hnb) {
  int i = blockIdx.x * blockDim.x + threadIdx.x;
  if (i >= 2 * BB * HH) return;
  hst[i] = 0.0f;
  cst[i] = 0.0f;
  hnb[i] = (bf16_t)0.0f;
  hnb[i + 2 * BB * HH] = (bf16_t)0.0f;
}

// ---------------------------------------------------------------------------
// One LSTM time step, both directions of one layer.
// Grid: (HH/16/8 = 4, 2 dirs), block 256 (8 waves).
// Wave 'slice' owns hidden units [hh0, hh0+16): computes 4 gate tiles x 2
// M-tiles (all 32 batch rows) with WMMA, then fuses gates in registers.
// ---------------------------------------------------------------------------
__global__ __launch_bounds__(256)
void lstm_step_kernel(const float* __restrict__ gx,        // [2][T*B][G]
                      const bf16_t* __restrict__ whh0, const bf16_t* __restrict__ whh1,
                      const float* __restrict__ noise0, const float* __restrict__ noise1,
                      const float* __restrict__ mask,      // [B,T]
                      float* __restrict__ hstate, float* __restrict__ cstate, // [2][B*H]
                      bf16_t* __restrict__ hnb,            // [2 parity][2 dir][B*H]
                      float* __restrict__ outF,            // layer1: d_out [B,T,2H] (else null)
                      bf16_t* __restrict__ outB,           // layer0: inter [T*B,2H] bf16 (else null)
                      int t) {
  const int tid  = threadIdx.x;
  const int lane = tid & 31;
  const int wave = tid >> 5;
  const int dir  = blockIdx.y;
  const int hh0  = (blockIdx.x * 8 + wave) * 16;

  const bf16_t* W     = dir ? whh1 : whh0;
  const float*  noise = dir ? noise1 : noise0;
  const float*  gxd   = gx + (size_t)dir * TT * BB * GG + (size_t)t * BB * GG;
  float* hs = hstate + dir * BB * HH;
  float* cs = cstate + dir * BB * HH;

  const int rp = (t + 1) & 1;   // parity written by step t-1 (zero-inited for t=0)
  const int wp = t & 1;
  const bf16_t* hn_rd = hnb + (size_t)rp * 2 * BB * HH + (size_t)dir * BB * HH;
  bf16_t*       hn_wr = hnb + (size_t)wp * 2 * BB * HH + (size_t)dir * BB * HH;

  const int t_act = dir ? (TT - 1 - t) : t;

  v8f acc[4][2];
#pragma unroll
  for (int g = 0; g < 4; ++g) { acc[g][0] = vzero8(); acc[g][1] = vzero8(); }

  for (int k0 = 0; k0 < HH; k0 += 32) {
    v16bf a0 = load_a_frag(hn_rd, HH, (lane & 15),      k0, lane);
    v16bf a1 = load_a_frag(hn_rd, HH, 16 + (lane & 15), k0, lane);
#pragma unroll
    for (int g = 0; g < 4; ++g) {
      v16bf bf = load_b_frag(W, HH, g * HH + hh0, k0, lane);
      acc[g][0] = wmma_bf16(a0, bf, acc[g][0]);
      acc[g][1] = wmma_bf16(a1, bf, acc[g][1]);
    }
  }

  const int nl = lane & 15, hb = lane >> 4;
  const int nn = hh0 + nl;  // hidden unit index
#pragma unroll
  for (int mt = 0; mt < 2; ++mt) {
#pragma unroll
    for (int r = 0; r < 8; ++r) {
      const int b = mt * 16 + hb * 8 + r;
      const size_t rowg = (size_t)b * GG;
      const float iv = acc[0][mt][r] + gxd[rowg + 0 * HH + nn];
      const float fv = acc[1][mt][r] + gxd[rowg + 1 * HH + nn];
      const float gv = acc[2][mt][r] + gxd[rowg + 2 * HH + nn];
      const float ov = acc[3][mt][r] + gxd[rowg + 3 * HH + nn];
      const float si = 1.0f / (1.0f + __expf(-iv));
      const float sf = 1.0f / (1.0f + __expf(-fv));
      const float so = 1.0f / (1.0f + __expf(-ov));
      const float tg = tanhf(gv);
      const size_t sidx = (size_t)b * HH + nn;
      const float c_old = cs[sidx], h_old = hs[sidx];
      const float c_n = sf * c_old + si * tg;
      const float h_n = so * tanhf(c_n);
      const float mv = mask[b * TT + t_act];
      const float h = h_n * mv + h_old * (1.0f - mv);
      const float c = c_n * mv + c_old * (1.0f - mv);
      hs[sidx] = h;
      cs[sidx] = c;
      hn_wr[sidx] = (bf16_t)(h * noise[sidx]);
      if (outF) outF[((size_t)b * TT + t_act) * (2 * HH) + dir * HH + nn] = h;
      else      outB[((size_t)t_act * BB + b) * (2 * HH) + dir * HH + nn] = (bf16_t)h;
    }
  }
}

// Append final h/c states into d_out tails: h_all then c_all, cell = layer*2+dir.
__global__ void copy_hc_kernel(const float* __restrict__ hst, const float* __restrict__ cst,
                               float* __restrict__ dout, int layer) {
  int i = blockIdx.x * blockDim.x + threadIdx.x;   // 0 .. 2*B*H-1
  if (i >= 2 * BB * HH) return;
  const int dir = i / (BB * HH);
  const int idx = i % (BB * HH);
  const int cell = layer * 2 + dir;
  const size_t OUT0 = (size_t)BB * TT * 2 * HH;
  dout[OUT0 + (size_t)cell * BB * HH + idx] = hst[i];
  dout[OUT0 + (size_t)4 * BB * HH + (size_t)cell * BB * HH + idx] = cst[i];
}

// ---------------------------------------------------------------------------
// Host launcher
// ---------------------------------------------------------------------------
extern "C" void kernel_launch(void* const* d_in, const int* in_sizes, int n_in,
                              void* d_out, int out_size, void* d_ws, size_t ws_size,
                              hipStream_t stream) {
  (void)in_sizes; (void)n_in; (void)out_size; (void)ws_size;
  const float* x    = (const float*)d_in[0];
  const float* mask = (const float*)d_in[1];
  const float* w_ih[4]  = {(const float*)d_in[2],  (const float*)d_in[7],
                           (const float*)d_in[12], (const float*)d_in[17]};
  const float* w_hh[4]  = {(const float*)d_in[3],  (const float*)d_in[8],
                           (const float*)d_in[13], (const float*)d_in[18]};
  const float* b_ih[4]  = {(const float*)d_in[4],  (const float*)d_in[9],
                           (const float*)d_in[14], (const float*)d_in[19]};
  const float* b_hh[4]  = {(const float*)d_in[5],  (const float*)d_in[10],
                           (const float*)d_in[15], (const float*)d_in[20]};
  const float* noise[4] = {(const float*)d_in[6],  (const float*)d_in[11],
                           (const float*)d_in[16], (const float*)d_in[21]};
  float* out = (float*)d_out;

  // Workspace layout (~325 MB total):
  //   bf16: xb[T*B*D] | inter[T*B*2H] | wihb[6.29M] | whhb[4.19M] | hnb[2*2*B*H]
  //   f32 : gx[2*T*B*G] | hstate[2*B*H] | cstate[2*B*H]
  char* ws = (char*)d_ws;
  bf16_t* xb    = (bf16_t*)ws;
  bf16_t* inter = xb    + (size_t)TT * BB * DD;            //  8,388,608 elems
  bf16_t* wihb  = inter + (size_t)TT * BB * 2 * HH;        // 16,777,216 elems
  bf16_t* whhb  = wihb  + (size_t)6291456;
  bf16_t* hnb   = whhb  + (size_t)4194304;
  float*  gx    = (float*)(ws + (size_t)71434240);         // bf16 region = 71,434,240 B
  float*  hst   = gx + (size_t)2 * TT * BB * GG;
  float*  cst   = hst + 2 * BB * HH;

  const size_t wih_off[4] = {0, 1048576, 2097152, 4194304};
  const int    wih_n[4]   = {1048576, 1048576, 2097152, 2097152};

  // --- convert weights and x to bf16 ---
  for (int i = 0; i < 4; ++i) {
    cvt_kernel<<<(wih_n[i] + 255) / 256, 256, 0, stream>>>(w_ih[i], wihb + wih_off[i], wih_n[i]);
    cvt_kernel<<<(1048576 + 255) / 256, 256, 0, stream>>>(w_hh[i], whhb + (size_t)i * 1048576, 1048576);
  }
  cvt_x_kernel<<<(BB * TT * DD + 255) / 256, 256, 0, stream>>>(x, xb);

  for (int layer = 0; layer < 2; ++layer) {
    const bf16_t* A = (layer == 0) ? xb : inter;
    const int K = (layer == 0) ? DD : 2 * HH;
    const int c0 = layer * 2, c1 = layer * 2 + 1;

    // Precompute gx for both directions of this layer (big WMMA GEMM).
    gemm_gx_kernel<<<dim3(GG / 128, (TT * BB) / (32 * 8), 2), 256, 0, stream>>>(
        A, wihb + wih_off[c0], wihb + wih_off[c1],
        b_ih[c0], b_hh[c0], b_ih[c1], b_hh[c1], gx, K);

    lstm_init_kernel<<<(2 * BB * HH + 255) / 256, 256, 0, stream>>>(hst, cst, hnb);

    float*  outF = (layer == 1) ? out : nullptr;
    bf16_t* outB = (layer == 0) ? inter : nullptr;
    for (int t = 0; t < TT; ++t) {
      lstm_step_kernel<<<dim3(HH / 16 / 8, 2), 256, 0, stream>>>(
          gx, whhb + (size_t)c0 * 1048576, whhb + (size_t)c1 * 1048576,
          noise[c0], noise[c1], mask, hst, cst, hnb, outF, outB, t);
    }
    copy_hc_kernel<<<(2 * BB * HH + 255) / 256, 256, 0, stream>>>(hst, cst, out, layer);
  }
}
